// HGT_policy_73065983640139
// MI455X (gfx1250) — matrix operand order repaired and checked by reading, not compiled, not run on hardware
//
#include <hip/hip_runtime.h>
#include <hip/hip_bf16.h>
#include <math.h>

// ---------------------------------------------------------------------------
// CDNA5 (gfx1250) HANConv forward.  wave32, WMMA 16x16x32 bf16 (fp32 acc),
// async global->LDS staging, fused per-head attention dots in GEMM epilogue.
// ---------------------------------------------------------------------------

#if defined(__has_builtin)
#  if __has_builtin(__builtin_amdgcn_global_load_async_to_lds_b128)
#    define HAVE_ASYNC_COPY 1
#  else
#    define HAVE_ASYNC_COPY 0
#  endif
#  if __has_builtin(__builtin_amdgcn_s_wait_asynccnt)
#    define HAVE_WAIT_ASYNC 1
#  else
#    define HAVE_WAIT_ASYNC 0
#  endif
#else
#  define HAVE_ASYNC_COPY 0
#  define HAVE_WAIT_ASYNC 0
#endif

typedef __attribute__((ext_vector_type(16))) __bf16 v16bf;
typedef __attribute__((ext_vector_type(8)))  float  v8f;
typedef __attribute__((ext_vector_type(8)))  unsigned int v8u;
typedef __attribute__((ext_vector_type(4)))  int    v4i;

// address-space-qualified 128-bit pointers for the async copy builtin
typedef __attribute__((address_space(1))) v4i* gv4p;  // global
typedef __attribute__((address_space(3))) v4i* lv4p;  // LDS

// fragment: 16 bf16 = 32 bytes = two 128-bit halves (forces ds_load_b128)
union Frag16 {
  v16bf b;
  v8u   u;
  struct { v4i lo, hi; } q;
};

__device__ __forceinline__ unsigned short f2bf(float f) {
  unsigned int x = __float_as_uint(f);
  unsigned int r = (x + 0x7FFFu + ((x >> 16) & 1u)) >> 16;  // RNE
  return (unsigned short)r;
}

__device__ __forceinline__ void atomicMaxF(float* addr, float v) {
  if (v >= 0.f) atomicMax((int*)addr, __float_as_int(v));
  else          atomicMin((unsigned int*)addr, __float_as_uint(v));
}

__device__ __forceinline__ void wait_async0() {
#if HAVE_WAIT_ASYNC
  __builtin_amdgcn_s_wait_asynccnt(0);
#else
  asm volatile("s_wait_asynccnt 0" ::: "memory");
#endif
}

// ---------------------------------------------------------------------------
// WMMA GEMM:  Y[M,128] = act(X[M,K] @ W[K,128] + bias)
//   mode 0: store (bias)                  mode 1: store (bias + relu)
//   mode 2: colsum += sum_rows tanh(.)    (semantic-attention fusion, no store)
//   in_relu: relu X while staging
//   natt/av*/ao*: fused per-(row,head) attention dots (mode 0 only):
//       ao[row*8+h] = sum_k (X@W+b)[row, h*16+k] * av[h*16+k]
// Block: 128 threads = 4 waves; each wave owns a 16-row x 128-col strip.
// LDS: sBuf (32KB) = fp32 X staging, later reused as bf16 W^T; sX (16KB) bf16.
// ---------------------------------------------------------------------------
__global__ __launch_bounds__(128) void wmma_gemm128_kernel(
    const float* __restrict__ X, int M, int K,
    const float* __restrict__ W, const float* __restrict__ bias,
    float* __restrict__ Y, float* __restrict__ colsum,
    int mode, int in_relu,
    const float* __restrict__ av0, const float* __restrict__ av1,
    const float* __restrict__ av2, const float* __restrict__ av3,
    float* __restrict__ ao0, float* __restrict__ ao1,
    float* __restrict__ ao2, float* __restrict__ ao3, int natt)
{
  __shared__ __attribute__((aligned(16))) unsigned int   sBuf[64 * 128];
  __shared__ __attribute__((aligned(16))) unsigned short sX[64 * 128];

  float* sXf = (float*)sBuf;                  // fp32 X staging view
  unsigned short* sW = (unsigned short*)sBuf; // bf16 W^T view ([col][K])

  const int tid = threadIdx.x;
  const int rowBase = blockIdx.x * 64;
  int validRows = M - rowBase; if (validRows > 64) validRows = 64;

#if HAVE_ASYNC_COPY
  {
    // async DMA the fp32 X tile into LDS (16B granules; rows are 16B-multiple)
    gv4p gsrc = (gv4p)(unsigned long long)(const void*)(X + (size_t)rowBase * K);
    lv4p ldst = (lv4p)sXf;
    const int ntrans = (validRows * K) >> 2;  // # of 16B transfers
    for (int t = tid; t < ntrans; t += 128) {
      __builtin_amdgcn_global_load_async_to_lds_b128(gsrc + t, ldst + t, 0, 0);
    }
    wait_async0();
    __syncthreads();
    for (int i = tid; i < 64 * K; i += 128) {
      int r = i / K;
      float f = (r < validRows) ? sXf[i] : 0.f;
      if (in_relu) f = f > 0.f ? f : 0.f;
      sX[i] = f2bf(f);
    }
    __syncthreads();  // sBuf now free to hold W^T
  }
#else
  for (int i = tid; i < 64 * K; i += 128) {
    int r = i / K, k = i - r * K;
    float f = (r < validRows) ? X[(size_t)(rowBase + r) * K + k] : 0.f;
    if (in_relu) f = f > 0.f ? f : 0.f;
    sX[i] = f2bf(f);
  }
#endif

  // stage W^T as bf16: sW[col*K + k] = W[k,col]
  for (int i = tid; i < 128 * K; i += 128) {
    int c = i / K, k = i - c * K;
    sW[c * K + k] = f2bf(W[k * 128 + c]);
  }
  __syncthreads();

  const int wave   = tid >> 5;
  const int lane   = tid & 31;
  const int laneHi = lane >> 4;
  const int lane16 = lane & 15;
  const int rowTile = wave * 16;

  v8f acc[8];
  const v8f vz = {0.f,0.f,0.f,0.f,0.f,0.f,0.f,0.f};
#pragma unroll
  for (int n = 0; n < 8; n++) acc[n] = vz;

  for (int kk = 0; kk < K; kk += 32) {
    // A fragment (16x32 bf16): per lane two contiguous 16B chunks
    //   u[v<4]  -> K = 2v+pos + laneHi*8        (shorts [base .. base+7])
    //   u[v>=4] -> K = 16+2(v-4)+pos + laneHi*8 (shorts [base+16 .. base+23])
    Frag16 fa;
    {
      const unsigned short* abase = &sX[(rowTile + lane16) * K + kk + laneHi * 8];
      fa.q.lo = *(const v4i*)abase;
      fa.q.hi = *(const v4i*)(abase + 16);
    }
#pragma unroll
    for (int n = 0; n < 8; n++) {
      // B fragment (32x16 bf16): two 16B chunks at column n*16+lane16
      //   u[v<4] -> K = 2v+pos + laneHi*8; u[v>=4] -> K = 16+2(v-4)+pos + laneHi*8
      Frag16 fb;
      {
        const unsigned short* bbase =
            &sW[(n * 16 + lane16) * K + kk + laneHi * 8];
        fb.q.lo = *(const v4i*)bbase;
        fb.q.hi = *(const v4i*)(bbase + 16);
      }
      acc[n] = __builtin_amdgcn_wmma_f32_16x16x32_bf16(
          false, fa.b, false, fb.b, (short)0, acc[n], false, false);
    }
  }

  // epilogue: C/D layout -> VGPR r: lanes 0-15 row M=r, lanes 16-31 row M=r+8
  if (mode != 2) {
    const float* avs[4] = {av0, av1, av2, av3};
    float*       aos[4] = {ao0, ao1, ao2, ao3};
#pragma unroll
    for (int n = 0; n < 8; n++) {
      int col = n * 16 + lane16;
      float bv = bias[col];
#pragma unroll
      for (int r = 0; r < 8; r++) {
        int row = rowBase + rowTile + r + laneHi * 8;
        float val = acc[n][r] + bv;
        if (mode == 1) val = val > 0.f ? val : 0.f;
        if (row < M) Y[(size_t)row * 128 + col] = val;
        // fused attention dots: head h == n, contract over lane16 (k)
#pragma unroll
        for (int v = 0; v < 4; v++) {
          if (v < natt) {
            float s = val * avs[v][col];
            s += __shfl_xor(s, 1);
            s += __shfl_xor(s, 2);
            s += __shfl_xor(s, 4);
            s += __shfl_xor(s, 8);
            if (lane16 == 0 && row < M) aos[v][(size_t)row * 8 + n] = s;
          }
        }
      }
    }
  } else {
#pragma unroll
    for (int n = 0; n < 8; n++) {
      int col = n * 16 + lane16;
      float bv = bias[col];
      float local = 0.f;
#pragma unroll
      for (int r = 0; r < 8; r++) {
        int row = rowBase + rowTile + r + laneHi * 8;
        if (row < M) local += tanhf(acc[n][r] + bv);
      }
      atomicAdd(&colsum[col], local);
    }
  }
}

__global__ __launch_bounds__(256) void fill_kernel(float* __restrict__ p,
                                                   float v, size_t n) {
  size_t i = (size_t)blockIdx.x * 256 + threadIdx.x;
  if (i < n) p[i] = v;
}

__device__ __forceinline__ float lrelu02(float a) {
  return a > 0.f ? a : 0.2f * a;
}

// segment-max of leaky_relu(a_s[s]+a_d[d]) per (dst, head)
__global__ __launch_bounds__(256) void edge_max_kernel(
    const int* __restrict__ ei, int E,
    const float* __restrict__ as_, const float* __restrict__ ad_,
    float* __restrict__ amax)
{
  int i = blockIdx.x * 256 + threadIdx.x;
  if (i >= E * 8) return;
  int e = i >> 3, h = i & 7;
  int s = ei[e], d = ei[E + e];
  float a = lrelu02(as_[s * 8 + h] + ad_[d * 8 + h]);
  atomicMaxF(&amax[d * 8 + h], a);
}

// segment-sum of exp(alpha - max)
__global__ __launch_bounds__(256) void edge_sum_kernel(
    const int* __restrict__ ei, int E,
    const float* __restrict__ as_, const float* __restrict__ ad_,
    const float* __restrict__ amax, float* __restrict__ den)
{
  int i = blockIdx.x * 256 + threadIdx.x;
  if (i >= E * 8) return;
  int e = i >> 3, h = i & 7;
  int s = ei[e], d = ei[E + e];
  float a = lrelu02(as_[s * 8 + h] + ad_[d * 8 + h]);
  atomicAdd(&den[d * 8 + h], __expf(a - amax[d * 8 + h]));
}

// weighted scatter: agg[d,:] += P[s,:] * softmax_weight(e,h); 1 wave / edge
__global__ __launch_bounds__(256) void edge_scatter_kernel(
    const int* __restrict__ ei, int E,
    const float* __restrict__ as_, const float* __restrict__ ad_,
    const float* __restrict__ amax, const float* __restrict__ den,
    const float* __restrict__ Psrc, float* __restrict__ agg)
{
  int t = blockIdx.x * 256 + threadIdx.x;
  int e = t >> 5;
  if (e >= E) return;
  int lane = t & 31;
  int s = ei[e], d = ei[E + e];
  __builtin_prefetch(&Psrc[(size_t)s * 128 + lane * 4], 0, 3);  // global_prefetch_b8
#pragma unroll
  for (int j = 0; j < 4; j++) {
    int ch = lane + 32 * j;
    int h = ch >> 4;
    float a = lrelu02(as_[s * 8 + h] + ad_[d * 8 + h]);
    float w = __expf(a - amax[d * 8 + h]) / (den[d * 8 + h] + 1e-16f);
    atomicAdd(&agg[(size_t)d * 128 + ch], Psrc[(size_t)s * 128 + ch] * w);
  }
}

// semantic attention over T=2 metapaths (single thread; trivial work)
__global__ void semantic_attn_kernel(const float* __restrict__ colsum,
                                     const float* __restrict__ q, float invN,
                                     float* __restrict__ attn)
{
  float s0 = 0.f, s1 = 0.f;
  for (int c = 0; c < 128; c++) {
    s0 += q[c] * colsum[c] * invN;
    s1 += q[c] * colsum[128 + c] * invN;
  }
  float m = fmaxf(s0, s1);
  float e0 = __expf(s0 - m), e1 = __expf(s1 - m);
  float dn = e0 + e1;
  attn[0] = e0 / dn;
  attn[1] = e1 / dn;
}

// in-place: dst = attn0*relu(dst) + attn1*relu(src1)
__global__ __launch_bounds__(256) void wsum2_kernel(
    float* __restrict__ dst, const float* __restrict__ src1,
    const float* __restrict__ attn, size_t n)
{
  size_t i = (size_t)blockIdx.x * 256 + threadIdx.x;
  if (i >= n) return;
  float a = dst[i];  a = a > 0.f ? a : 0.f;
  float b = src1[i]; b = b > 0.f ? b : 0.f;
  dst[i] = attn[0] * a + attn[1] * b;
}

__global__ __launch_bounds__(256) void relu_inplace_kernel(float* __restrict__ p,
                                                           size_t n) {
  size_t i = (size_t)blockIdx.x * 256 + threadIdx.x;
  if (i < n) { float v = p[i]; p[i] = v > 0.f ? v : 0.f; }
}

// column sums for mean pooling (64 rows / block, 2 threads / column)
__global__ __launch_bounds__(256) void colsum_pool_kernel(
    const float* __restrict__ X, int M, float* __restrict__ out)
{
  int col = threadIdx.x & 127;
  int sub = threadIdx.x >> 7;
  int base = blockIdx.x * 64;
  float acc = 0.f;
  for (int r = sub; r < 64; r += 2) {
    int row = base + r;
    if (row < M) acc += X[(size_t)row * 128 + col];
  }
  atomicAdd(&out[col], acc);
}

// shared MLP + 3 policy heads + value head, single block (tiny)
__global__ __launch_bounds__(128) void head_kernel(
    const float* __restrict__ pooled, float invN,
    const float* w0, const float* b0, const float* w1, const float* b1,
    const float* p0w1, const float* p0b1, const float* p0w2, const float* p0b2,
    const float* p1w1, const float* p1b1, const float* p1w2, const float* p1b2,
    const float* p2w1, const float* p2b1, const float* p2w2, const float* p2b2,
    const float* vw1, const float* vb1, const float* vw2, const float* vb2,
    float* __restrict__ out)
{
  __shared__ float s0[256], s1[128], s2[128], t64[64];
  const int tid = threadIdx.x;
  for (int i = tid; i < 256; i += 128) s0[i] = pooled[i] * invN;
  __syncthreads();
  { float a = b0[tid];
    for (int k = 0; k < 256; k++) a += s0[k] * w0[k * 128 + tid];
    s1[tid] = fmaxf(a, 0.f); }
  __syncthreads();
  { float a = b1[tid];
    for (int k = 0; k < 128; k++) a += s1[k] * w1[k * 128 + tid];
    s2[tid] = fmaxf(a, 0.f); }
  __syncthreads();

  if (tid < 64) { float a = p0b1[tid];
    for (int k = 0; k < 128; k++) a += s2[k] * p0w1[k * 64 + tid];
    t64[tid] = fmaxf(a, 0.f); }
  __syncthreads();
  if (tid < 5) { float a = p0b2[tid];
    for (int k = 0; k < 64; k++) a += t64[k] * p0w2[k * 5 + tid];
    out[tid] = a; }
  __syncthreads();
  if (tid < 64) { float a = p1b1[tid];
    for (int k = 0; k < 128; k++) a += s2[k] * p1w1[k * 64 + tid];
    t64[tid] = fmaxf(a, 0.f); }
  __syncthreads();
  if (tid < 5) { float a = p1b2[tid];
    for (int k = 0; k < 64; k++) a += t64[k] * p1w2[k * 5 + tid];
    out[5 + tid] = a; }
  __syncthreads();
  if (tid < 64) { float a = p2b1[tid];
    for (int k = 0; k < 128; k++) a += s2[k] * p2w1[k * 64 + tid];
    t64[tid] = fmaxf(a, 0.f); }
  __syncthreads();
  if (tid < 10) { float a = p2b2[tid];
    for (int k = 0; k < 64; k++) a += t64[k] * p2w2[k * 10 + tid];
    out[10 + tid] = a; }
  __syncthreads();
  if (tid < 64) { float a = vb1[tid];
    for (int k = 0; k < 128; k++) a += s2[k] * vw1[k * 64 + tid];
    t64[tid] = fmaxf(a, 0.f); }
  __syncthreads();
  if (tid == 0) { float a = vb2[0];
    for (int k = 0; k < 64; k++) a += t64[k] * vw2[k];
    out[20] = a; }
}

// ---------------------------------------------------------------------------
static inline int cdiv(long long a, long long b) { return (int)((a + b - 1) / b); }

extern "C" void kernel_launch(void* const* d_in, const int* in_sizes, int n_in,
                              void* d_out, int out_size, void* d_ws, size_t ws_size,
                              hipStream_t stream)
{
  const float* x_a = (const float*)d_in[0];
  const float* x_b = (const float*)d_in[1];
  const int* e_ab = (const int*)d_in[2];
  const int* e_ba = (const int*)d_in[3];
  const int* e_aa = (const int*)d_in[4];
  const int N = in_sizes[0] / 64;
  const int E = in_sizes[2] / 2;

  int idx = 5;
  #define PP() ((const float*)d_in[idx++])
  const float* inA_w = PP(); const float* inA_b = PP();
  const float* inB_w = PP(); const float* inB_b = PP();
  struct LayerP {
    const float *pAw, *pAb, *pBw, *pBb;
    const float *asrc[3], *adst[3];
    const float *kw, *kb, *q;
  } L[2];
  for (int l = 0; l < 2; l++) {
    L[l].pAw = PP(); L[l].pAb = PP(); L[l].pBw = PP(); L[l].pBb = PP();
    for (int t = 0; t < 3; t++) L[l].asrc[t] = PP();
    for (int t = 0; t < 3; t++) L[l].adst[t] = PP();
    L[l].kw = PP(); L[l].kb = PP(); L[l].q = PP();
  }
  const float* sw0 = PP(); const float* sb0 = PP();
  const float* sw1 = PP(); const float* sb1 = PP();
  const float* pw[3][2]; const float* pb[3][2];
  for (int hdim = 0; hdim < 3; hdim++) {
    pw[hdim][0] = PP(); pb[hdim][0] = PP();
    pw[hdim][1] = PP(); pb[hdim][1] = PP();
  }
  const float* vw1 = PP(); const float* vb1 = PP();
  const float* vw2 = PP(); const float* vb2 = PP();
  #undef PP

  // ---- workspace layout ----
  float* w = (float*)d_ws;
  const size_t nf128 = (size_t)N * 128;
  const size_t nf8 = (size_t)N * 8;
  float* hA    = w; w += nf128;   // h["a"]; agg for edge "ba"; new h["a"]
  float* hB    = w; w += nf128;   // h["b"]; agg for edge "ab"; new h["b"]
  float* pA    = w; w += nf128;   // projected a
  float* pB    = w; w += nf128;   // projected b
  float* aggAA = w; w += nf128;   // agg for edge "aa"
  float* aA0 = w; w += nf8;       // pA . att_src[ab]
  float* aA1 = w; w += nf8;       // pA . att_src[aa]
  float* aA2 = w; w += nf8;       // pA . att_dst[ba]
  float* aA3 = w; w += nf8;       // pA . att_dst[aa]
  float* aB0 = w; w += nf8;       // pB . att_dst[ab]
  float* aB1 = w; w += nf8;       // pB . att_src[ba]
  float* amax  = w; w += nf8;
  float* den   = w; w += nf8;
  float* cs2   = w; w += 256;
  float* attn  = w; w += 2;
  float* pooled = w; w += 256;

  const int gemmGrid = cdiv(N, 64);
  const int nhGrid = cdiv((long long)N * 8, 256);
  const int e8Grid = cdiv((long long)E * 8, 256);
  const int scatGrid = cdiv((long long)E * 32, 256);
  const int elemGrid = cdiv((long long)N * 128, 256);
  const float invN = 1.0f / (float)N;

  #define NOATT nullptr, nullptr, nullptr, nullptr, nullptr, nullptr, nullptr, nullptr, 0

  // ---- input projections: h = relu(x @ W + b) ----
  wmma_gemm128_kernel<<<gemmGrid, 128, 0, stream>>>(x_a, N, 64, inA_w, inA_b,
                                                    hA, nullptr, 1, 0, NOATT);
  wmma_gemm128_kernel<<<gemmGrid, 128, 0, stream>>>(x_b, N, 32, inB_w, inB_b,
                                                    hB, nullptr, 1, 0, NOATT);

  for (int l = 0; l < 2; l++) {
    // node-type projections with fused per-head attention dots
    wmma_gemm128_kernel<<<gemmGrid, 128, 0, stream>>>(
        hA, N, 128, L[l].pAw, L[l].pAb, pA, nullptr, 0, 0,
        L[l].asrc[0], L[l].asrc[2], L[l].adst[1], L[l].adst[2],
        aA0, aA1, aA2, aA3, 4);
    wmma_gemm128_kernel<<<gemmGrid, 128, 0, stream>>>(
        hB, N, 128, L[l].pBw, L[l].pBb, pB, nullptr, 0, 0,
        L[l].adst[0], L[l].asrc[1], nullptr, nullptr,
        aB0, aB1, nullptr, nullptr, 2);

    // edge passes: ab (a->b, agg in hB), ba (b->a, agg in hA), aa (aggAA)
    struct ER { const int* ei; const float* sp; const float* as_; const float* ad_; float* agg; };
    ER er[3] = { {e_ab, pA, aA0, aB0, hB},
                 {e_ba, pB, aB1, aA2, hA},
                 {e_aa, pA, aA1, aA3, aggAA} };
    for (int k = 0; k < 3; k++) {
      fill_kernel<<<nhGrid, 256, 0, stream>>>(amax, -INFINITY, nf8);
      fill_kernel<<<nhGrid, 256, 0, stream>>>(den, 0.f, nf8);
      fill_kernel<<<elemGrid, 256, 0, stream>>>(er[k].agg, 0.f, nf128);
      edge_max_kernel<<<e8Grid, 256, 0, stream>>>(er[k].ei, E, er[k].as_, er[k].ad_, amax);
      edge_sum_kernel<<<e8Grid, 256, 0, stream>>>(er[k].ei, E, er[k].as_, er[k].ad_, amax, den);
      edge_scatter_kernel<<<scatGrid, 256, 0, stream>>>(er[k].ei, E, er[k].as_, er[k].ad_,
                                                        amax, den, er[k].sp, er[k].agg);
    }

    // semantic attention for type "a" (metapaths: ba -> hA, aa -> aggAA)
    fill_kernel<<<1, 256, 0, stream>>>(cs2, 0.f, (size_t)256);
    wmma_gemm128_kernel<<<gemmGrid, 128, 0, stream>>>(hA, N, 128, L[l].kw, L[l].kb,
                                                      nullptr, cs2, 2, 1, NOATT);
    wmma_gemm128_kernel<<<gemmGrid, 128, 0, stream>>>(aggAA, N, 128, L[l].kw, L[l].kb,
                                                      nullptr, cs2 + 128, 2, 1, NOATT);
    semantic_attn_kernel<<<1, 1, 0, stream>>>(cs2, L[l].q, invN, attn);
    wsum2_kernel<<<elemGrid, 256, 0, stream>>>(hA, aggAA, attn, nf128);
    // type "b": single metapath -> softmax weight 1 -> just relu
    relu_inplace_kernel<<<elemGrid, 256, 0, stream>>>(hB, nf128);
  }

  // ---- mean pooling + heads ----
  fill_kernel<<<1, 256, 0, stream>>>(pooled, 0.f, (size_t)256);
  const int poolGrid = cdiv(N, 64);
  colsum_pool_kernel<<<poolGrid, 256, 0, stream>>>(hA, N, pooled);
  colsum_pool_kernel<<<poolGrid, 256, 0, stream>>>(hB, N, pooled + 128);
  head_kernel<<<1, 128, 0, stream>>>(pooled, invN, sw0, sb0, sw1, sb1,
                                     pw[0][0], pb[0][0], pw[0][1], pb[0][1],
                                     pw[1][0], pb[1][0], pw[1][1], pb[1][1],
                                     pw[2][0], pb[2][0], pw[2][1], pb[2][1],
                                     vw1, vb1, vw2, vb2, (float*)d_out);
  #undef NOATT
}
